// SpatialViewModel_1090921693568
// MI455X (gfx1250) — matrix operand order repaired and checked by reading, not compiled, not run on hardware
//
#include <hip/hip_runtime.h>

typedef __attribute__((ext_vector_type(16))) __bf16 v16bf;
typedef __attribute__((ext_vector_type(8)))  float  v8f;
typedef __attribute__((ext_vector_type(4)))  float  f4;

#define GS0 1024
#define GS1 1024
#define FDIM 64

union AV { v16bf v; unsigned u[8]; };

// ---- RNE float->bf16 (used only in the one-shot V pack kernel) -------------
static __device__ __forceinline__ unsigned f2bf(float f) {
  unsigned u = __float_as_uint(f);
  unsigned r = u + 0x7FFFu + ((u >> 16) & 1u);
  return r >> 16;
}
static __device__ __forceinline__ float bf2f(unsigned h) {
  return __uint_as_float(h << 16);
}
static __device__ __forceinline__ void split_pack_rne(float a, float b,
                                                      unsigned& hi, unsigned& lo) {
  unsigned ha = f2bf(a), hb = f2bf(b);
  float ra = a - bf2f(ha);
  float rb = b - bf2f(hb);
  hi = ha | (hb << 16);
  lo = f2bf(ra) | (f2bf(rb) << 16);
}

// ---- cheap truncation-based split for the streaming side -------------------
// hi = {b[31:16], a[31:16]} in ONE v_perm_b32; residual a - trunc16(a) is
// EXACT in fp32, so hi+lo accuracy (~2^-16 rel) matches the RNE variant.
static __device__ __forceinline__ void split_pack2(float a, float b,
                                                   unsigned& hi, unsigned& lo) {
  unsigned ua = __float_as_uint(a), ub = __float_as_uint(b);
  hi = __builtin_amdgcn_perm(ub, ua, 0x07060302u);       // {b.hi16, a.hi16}
  float ra = a - __uint_as_float(ua & 0xFFFF0000u);      // exact residual
  float rb = b - __uint_as_float(ub & 0xFFFF0000u);
  lo = __builtin_amdgcn_perm(__float_as_uint(rb), __float_as_uint(ra),
                             0x07060302u);
}

// ---------------------------------------------------------------------------
// Kernel 1: V = x[sx,sy] @ W (64x64), packed into WMMA B-matrix layout
// (bf16 hi plane + residual-lo plane) in workspace.
// bpk: hi[ks][ntile][lane][dw] dwords (2*4*32*8 = 2048), lo plane at +2048.
//   lane n-col = ntile*16 + (lane&15)
//   K of dword dw = ks*32 + (dw>>2)*16 + (lane>>4)*8 + 2*(dw&3) (+1 in high half)
// ---------------------------------------------------------------------------
__global__ __launch_bounds__(256) void build_v_pack(
    const float* __restrict__ x, const float* __restrict__ W,
    const int* __restrict__ sx, const int* __restrict__ sy,
    unsigned* __restrict__ bpk) {
  __shared__ float Vs[FDIM][FDIM];
  const int tid = threadIdx.x;

  for (int t = 0; t < 16; ++t) {
    int e = tid + 256 * t;          // 0..4095
    int k = e >> 6;
    int f = e & 63;
    const float* xr = x + ((long)sx[k] * GS1 + (long)sy[k]) * FDIM;
    float acc = 0.0f;
#pragma unroll
    for (int c = 0; c < FDIM; ++c) acc += xr[c] * W[c * FDIM + f];
    Vs[k][f] = acc;
  }
  __syncthreads();

  for (int t = 0; t < 8; ++t) {
    int p = tid + 256 * t;          // 0..2047
    int dw    =  p & 7;
    int lane  = (p >> 3) & 31;
    int ntile = (p >> 8) & 3;
    int ks    =  p >> 10;
    int n  = ntile * 16 + (lane & 15);
    int k0 = ks * 32 + (dw >> 2) * 16 + (lane >> 4) * 8 + (dw & 3) * 2;
    unsigned hi, lo;
    split_pack_rne(Vs[k0][n], Vs[k0 + 1][n], hi, lo);
    bpk[p]        = hi;
    bpk[p + 2048] = lo;
  }
}

// ---------------------------------------------------------------------------
// Kernel 2: out[M,64] = P[M,64] @ V[64,64] via split-bf16 WMMA.
// 8 waves/block; each wave computes a 16x64 tile (4 n-tiles, K=64 = 2 steps).
// 3 bf16 WMMAs per K-step emulate fp32: Ph*Bh + Pl*Bh + Ph*Bl.
// ---------------------------------------------------------------------------
__global__ __launch_bounds__(256) void spatial_wmma(
    const float* __restrict__ P, const unsigned* __restrict__ bpk,
    float* __restrict__ out) {
  const int lane = threadIdx.x & 31;
  const int wv   = threadIdx.x >> 5;
  const long m0  = (long)blockIdx.x * 128 + (long)wv * 16;
  const int ln16 = lane & 15;       // row-in-tile for A, col for B/D
  const int hf   = lane >> 4;       // lane half: K sub-range / D row+8

  // A-matrix (16x32 bf16) lane layout per K-step ks:
  //   lanes 0-15 : VGPR0-3 = K 0..7,  VGPR4-7 = K 16..23   (+32*ks)
  //   lanes16-31 : VGPR0-3 = K 8..15, VGPR4-7 = K 24..31   (+32*ks)
  const float* prow = P + (m0 + ln16) * FDIM + hf * 8;

  AV aHi[2], aLo[2];
#pragma unroll
  for (int ks = 0; ks < 2; ++ks) {
    const f4* c0 = (const f4*)(prow + ks * 32);
    const f4* c1 = (const f4*)(prow + ks * 32 + 16);
    f4 q0 = __builtin_nontemporal_load(c0);
    f4 q1 = __builtin_nontemporal_load(c0 + 1);
    f4 q2 = __builtin_nontemporal_load(c1);
    f4 q3 = __builtin_nontemporal_load(c1 + 1);
    split_pack2(q0.x, q0.y, aHi[ks].u[0], aLo[ks].u[0]);
    split_pack2(q0.z, q0.w, aHi[ks].u[1], aLo[ks].u[1]);
    split_pack2(q1.x, q1.y, aHi[ks].u[2], aLo[ks].u[2]);
    split_pack2(q1.z, q1.w, aHi[ks].u[3], aLo[ks].u[3]);
    split_pack2(q2.x, q2.y, aHi[ks].u[4], aLo[ks].u[4]);
    split_pack2(q2.z, q2.w, aHi[ks].u[5], aLo[ks].u[5]);
    split_pack2(q3.x, q3.y, aHi[ks].u[6], aLo[ks].u[6]);
    split_pack2(q3.z, q3.w, aHi[ks].u[7], aLo[ks].u[7]);
  }

#pragma unroll
  for (int nt = 0; nt < 4; ++nt) {
    v8f acc = {};
#pragma unroll
    for (int ks = 0; ks < 2; ++ks) {
      // B planes: 8 dwords per lane, hi at p, lo at p + 2048 dwords
      const uint4* bp =
          (const uint4*)(bpk + (((ks * 4 + nt) * 32) + lane) * 8);
      uint4 h0 = bp[0], h1 = bp[1];
      uint4 l0 = bp[512], l1 = bp[513];
      AV bHi, bLo;
      bHi.u[0] = h0.x; bHi.u[1] = h0.y; bHi.u[2] = h0.z; bHi.u[3] = h0.w;
      bHi.u[4] = h1.x; bHi.u[5] = h1.y; bHi.u[6] = h1.z; bHi.u[7] = h1.w;
      bLo.u[0] = l0.x; bLo.u[1] = l0.y; bLo.u[2] = l0.z; bLo.u[3] = l0.w;
      bLo.u[4] = l1.x; bLo.u[5] = l1.y; bLo.u[6] = l1.z; bLo.u[7] = l1.w;
      acc = __builtin_amdgcn_wmma_f32_16x16x32_bf16(
          false, aHi[ks].v, false, bHi.v, (short)0, acc, false, false);
      acc = __builtin_amdgcn_wmma_f32_16x16x32_bf16(
          false, aLo[ks].v, false, bHi.v, (short)0, acc, false, false);
      acc = __builtin_amdgcn_wmma_f32_16x16x32_bf16(
          false, aHi[ks].v, false, bLo.v, (short)0, acc, false, false);
    }
    // D layout: VGPR r -> row (m0 + 8*hf + r), col = nt*16 + ln16
    float* orow = out + (m0 + hf * 8) * FDIM + nt * 16 + ln16;
#pragma unroll
    for (int r = 0; r < 8; ++r)
      __builtin_nontemporal_store(acc[r], &orow[(long)r * FDIM]);
  }
}

extern "C" void kernel_launch(void* const* d_in, const int* in_sizes, int n_in,
                              void* d_out, int out_size, void* d_ws, size_t ws_size,
                              hipStream_t stream) {
  (void)in_sizes; (void)n_in; (void)out_size; (void)ws_size;
  const float* x  = (const float*)d_in[0];
  const float* W  = (const float*)d_in[1];
  const float* P  = (const float*)d_in[2];
  const int*   sx = (const int*)d_in[3];
  const int*   sy = (const int*)d_in[4];
  float*    out = (float*)d_out;
  unsigned* bpk = (unsigned*)d_ws;   // 16 KB: bf16 B hi plane + lo plane

  build_v_pack<<<1, 256, 0, stream>>>(x, W, sx, sy, bpk);

  const long M = (long)GS0 * GS1;            // 1,048,576 rows
  const int blocks = (int)(M / 128);         // 8192 blocks x 128 rows
  spatial_wmma<<<blocks, 256, 0, stream>>>(P, bpk, out);
}
